// Attention_43662637531229
// MI455X (gfx1250) — matrix operand order repaired and checked by reading, not compiled, not run on hardware
//
#include <hip/hip_runtime.h>
#include <hip/hip_bf16.h>
#include <math.h>

// ---------------------------------------------------------------------------
// Problem constants (from reference): B=4, C=192, H=W=128, HEADS=8, C2=384
// ---------------------------------------------------------------------------
#define BATCH 4
#define CIN   192
#define CC2   384
#define HEADS 8
#define CH    24        // channels per head
#define CHP   32        // padded channels per head (WMMA K/M padding)
#define IMGH  128
#define IMGW  128
#define NPIX  16384     // 128*128

typedef __bf16 bf16_t;
typedef bf16_t v8bf  __attribute__((ext_vector_type(8)));
typedef bf16_t v16bf __attribute__((ext_vector_type(16)));
typedef float  v8f   __attribute__((ext_vector_type(8)));

#define CAT16(lo, hi) __builtin_shufflevector((lo), (hi), \
    0, 1, 2, 3, 4, 5, 6, 7, 8, 9, 10, 11, 12, 13, 14, 15)

// ---------------------------------------------------------------------------
// 1) weights f32 -> bf16
// ---------------------------------------------------------------------------
__global__ void cvt_weights(const float* __restrict__ kv_w,
                            const float* __restrict__ q_w,
                            const float* __restrict__ ky_w,
                            bf16_t* __restrict__ wbf)     // (3, C2, CIN)
{
    int idx = blockIdx.x * blockDim.x + threadIdx.x;
    const int total = 3 * CC2 * CIN;
    if (idx >= total) return;
    int t = idx / (CC2 * CIN);
    int o = idx % (CC2 * CIN);
    const float* src = (t == 0) ? kv_w : (t == 1) ? q_w : ky_w;
    wbf[idx] = (bf16_t)src[o];
}

// ---------------------------------------------------------------------------
// 2) inputs f32 (b, c, n) -> bf16 TRANSPOSED (b, n, c), via LDS 32x32 tile,
//    so conv B-fragments become contiguous 32B loads. Also fuses x_mid=x+y.
// ---------------------------------------------------------------------------
__global__ void cvt_inputs_t(const float* __restrict__ x,
                             const float* __restrict__ y,
                             bf16_t* __restrict__ xT,
                             bf16_t* __restrict__ yT,
                             bf16_t* __restrict__ xmT)
{
    __shared__ bf16_t tx[32][33], ty[32][33], tm[32][33];
    const int n0 = blockIdx.x * 32;
    const int c0 = blockIdx.y * 32;
    const int b  = blockIdx.z;
    const int tn = threadIdx.x & 31;
    const int tr = threadIdx.x >> 5;
#pragma unroll
    for (int i = 0; i < 4; ++i) {
        int cl = tr + 8 * i;
        size_t o = ((size_t)b * CIN + c0 + cl) * NPIX + n0 + tn;   // coalesced in tn
        float xv = x[o], yv = y[o];
        tx[cl][tn] = (bf16_t)xv;
        ty[cl][tn] = (bf16_t)yv;
        tm[cl][tn] = (bf16_t)(xv + yv);
    }
    __syncthreads();
#pragma unroll
    for (int i = 0; i < 4; ++i) {
        int nl = tr + 8 * i;
        size_t o = ((size_t)b * NPIX + n0 + nl) * CIN + c0 + tn;   // coalesced in tn
        xT[o]  = tx[tn][nl];
        yT[o]  = ty[tn][nl];
        xmT[o] = tm[tn][nl];
    }
}

// ---------------------------------------------------------------------------
// 3) conv1x1 as WMMA GEMM: Out[b, co, p] = sum_ci W[co,ci]*InT[b,p,ci] + bias
//    Block: 16 (co) x 128 (p), 8 waves; W tile staged in LDS; K unrolled (6).
// ---------------------------------------------------------------------------
__global__ void conv1x1_wmma(const bf16_t* __restrict__ inT,  // (B, NPIX, CIN)
                             const bf16_t* __restrict__ wbf,  // (C2, CIN)
                             const float*  __restrict__ bias, // (C2)
                             float* __restrict__ out)         // (B, C2, NPIX)
{
    __shared__ bf16_t a_lds[16 * CIN];

    const int bidx = blockIdx.z;
    const int co0  = blockIdx.y * 16;
    const int wave = threadIdx.x >> 5;
    const int l    = threadIdx.x & 31;
    const int n0   = (blockIdx.x * 8 + wave) * 16;
    const int m    = l & 15;
    const int g    = l >> 4;

    for (int i = threadIdx.x; i < 16 * CIN; i += blockDim.x) {
        int mm = i / CIN, kk = i % CIN;
        a_lds[i] = wbf[(co0 + mm) * CIN + kk];
    }
    __syncthreads();

    // per-lane bases: A row in LDS, B column (pixel n0+m) in transposed input
    const bf16_t* arow = a_lds + m * CIN;
    const bf16_t* bcol = inT + ((size_t)bidx * NPIX + n0 + m) * CIN;

    v8f acc = {};
#pragma unroll
    for (int k0 = 0; k0 < CIN; k0 += 32) {
        const v8bf* pa = (const v8bf*)(arow + k0);            // 16B LDS loads
        v16bf a = CAT16(pa[2 * g], pa[2 * g + 1]);            // K = k0+16g+e
        v16bf b = *(const v16bf*)(bcol + k0 + 16 * g);        // K = k0+16g+e, 32B
        acc = __builtin_amdgcn_wmma_f32_16x16x32_bf16(
            false, a, false, b, (short)0, acc, false, false);
    }

    float* outb = out + ((size_t)bidx * CC2 + co0) * NPIX;
#pragma unroll
    for (int r = 0; r < 8; ++r) {
        int row = r + 8 * g;
        outb[(size_t)row * NPIX + n0 + m] = acc[r] + bias[co0 + row];
    }
}

// ---------------------------------------------------------------------------
// 4) grouped 3x3 conv (groups=192, 2 in / 2 out per group), f32 stencil
// ---------------------------------------------------------------------------
__global__ void gconv3(const float* __restrict__ in,   // (B, C2, H, W)
                       const float* __restrict__ w,    // (C2, 2, 3, 3)
                       const float* __restrict__ bias, // (C2)
                       float* __restrict__ out)        // (B, C2, H, W)
{
    size_t idx = (size_t)blockIdx.x * blockDim.x + threadIdx.x;
    if (idx >= (size_t)BATCH * CC2 * NPIX) return;
    int xx0 = (int)(idx % IMGW);
    int yy0 = (int)((idx / IMGW) % IMGH);
    int co  = (int)((idx / NPIX) % CC2);
    int b   = (int)(idx / ((size_t)CC2 * NPIX));
    int grp = co >> 1;
    float s = bias[co];
#pragma unroll
    for (int j = 0; j < 2; ++j) {
        const float* ip = in + ((size_t)b * CC2 + (2 * grp + j)) * NPIX;
        const float* wp = w + ((co * 2 + j) * 9);
#pragma unroll
        for (int dy = -1; dy <= 1; ++dy) {
            int yy = yy0 + dy;
            if (yy < 0 || yy >= IMGH) continue;
#pragma unroll
            for (int dx = -1; dx <= 1; ++dx) {
                int xc = xx0 + dx;
                if (xc < 0 || xc >= IMGW) continue;
                s += wp[(dy + 1) * 3 + (dx + 1)] * ip[yy * IMGW + xc];
            }
        }
    }
    out[idx] = s;
}

// ---------------------------------------------------------------------------
// 5) L2 row norms over the pixel axis, 1/max(norm, 1e-12)
// ---------------------------------------------------------------------------
__global__ void rownorm(const float* __restrict__ gq,
                        const float* __restrict__ gkvx,
                        const float* __restrict__ gkvy,
                        float* __restrict__ invn)      // (3, B, C2)
{
    __shared__ float red[256];
    int rid = blockIdx.x;
    int c = rid % CC2;
    int b = (rid / CC2) % BATCH;
    int t = rid / (BATCH * CC2);
    const float* src = (t == 0) ? gq : (t == 1) ? gkvx : gkvy;
    const float* row = src + ((size_t)b * CC2 + c) * NPIX;
    float s = 0.f;
    for (int i = threadIdx.x; i < NPIX; i += blockDim.x) {
        float v = row[i];
        s += v * v;
    }
    red[threadIdx.x] = s;
    __syncthreads();
    for (int o = 128; o > 0; o >>= 1) {
        if ((int)threadIdx.x < o) red[threadIdx.x] += red[threadIdx.x + o];
        __syncthreads();
    }
    if (threadIdx.x == 0)
        invn[rid] = 1.f / fmaxf(sqrtf(red[0]), 1e-12f);
}

// ---------------------------------------------------------------------------
// 6) pack q1,q2,kx,ky (normalized) into padded (B, HEADS, 32, NPIX) bf16,
//    row-major along n (both GEMM-A chunks and GEMM-B runs are contiguous).
// ---------------------------------------------------------------------------
__global__ void pack_qk(const float* __restrict__ gq,
                        const float* __restrict__ gkvx,
                        const float* __restrict__ gkvy,
                        const float* __restrict__ invn,
                        bf16_t* __restrict__ q1n, bf16_t* __restrict__ q2n,
                        bf16_t* __restrict__ kxn, bf16_t* __restrict__ kyn)
{
    size_t idx = (size_t)blockIdx.x * blockDim.x + threadIdx.x;
    const size_t per_t = (size_t)BATCH * HEADS * CHP * NPIX;
    if (idx >= 4 * per_t) return;
    int t = (int)(idx / per_t);
    size_t o = idx % per_t;
    int n  = (int)(o % NPIX);
    int r  = (int)((o / NPIX) % CHP);
    int hd = (int)((o / ((size_t)NPIX * CHP)) % HEADS);
    int b  = (int)(o / ((size_t)NPIX * CHP * HEADS));

    float val = 0.f;
    if (r < CH) {
        int c = hd * CH + r;
        switch (t) {
            case 0: val = gq  [((size_t)b * CC2 + c)       * NPIX + n] * invn[(0 * BATCH + b) * CC2 + c];       break;
            case 1: val = gq  [((size_t)b * CC2 + CIN + c) * NPIX + n] * invn[(0 * BATCH + b) * CC2 + CIN + c]; break;
            case 2: val = gkvx[((size_t)b * CC2 + c)       * NPIX + n] * invn[(1 * BATCH + b) * CC2 + c];       break;
            default:val = gkvy[((size_t)b * CC2 + c)       * NPIX + n] * invn[(2 * BATCH + b) * CC2 + c];       break;
        }
    }
    bf16_t* dst = (t == 0) ? q1n : (t == 1) ? q2n : (t == 2) ? kxn : kyn;
    dst[o] = (bf16_t)val;
}

// ---------------------------------------------------------------------------
// 7) pack vx,vy TRANSPOSED into (B*HEADS, NPIX, 32) bf16 (pad channels 0),
//    via LDS tile so both global sides are coalesced. Makes out-GEMM
//    B-fragments single 32B contiguous loads.
// ---------------------------------------------------------------------------
__global__ void pack_v_t(const float* __restrict__ gkvx,
                         const float* __restrict__ gkvy,
                         bf16_t* __restrict__ vxT,
                         bf16_t* __restrict__ vyT)
{
    __shared__ bf16_t tvx[32][33], tvy[32][33];
    const int n0 = blockIdx.x * 32;
    const int bh = blockIdx.y;             // b*HEADS + hd
    const int hd = bh & 7, b = bh >> 3;
    const int tn = threadIdx.x & 31;
    const int tr = threadIdx.x >> 5;
#pragma unroll
    for (int i = 0; i < 4; ++i) {
        int r = tr + 8 * i;                // channel within head (0..31)
        float vx = 0.f, vy = 0.f;
        if (r < CH) {
            size_t o = ((size_t)b * CC2 + CIN + hd * CH + r) * NPIX + n0 + tn;
            vx = gkvx[o];
            vy = gkvy[o];
        }
        tvx[r][tn] = (bf16_t)vx;
        tvy[r][tn] = (bf16_t)vy;
    }
    __syncthreads();
#pragma unroll
    for (int i = 0; i < 4; ++i) {
        int nl = tr + 8 * i;
        size_t o = ((size_t)bh * NPIX + n0 + nl) * CHP + tn;   // coalesced in tn
        vxT[o] = tvx[tn][nl];
        vyT[o] = tvy[tn][nl];
    }
}

// ---------------------------------------------------------------------------
// 8) attn scores: S = q1 @ ky^T + q2 @ kx^T over n (K = 16384).
//    One block per (b,h); 4 waves = 2x2 tiles over padded 32x32.
// ---------------------------------------------------------------------------
__global__ void attn_gemm(const bf16_t* __restrict__ q1n,
                          const bf16_t* __restrict__ q2n,
                          const bf16_t* __restrict__ kxn,
                          const bf16_t* __restrict__ kyn,
                          float* __restrict__ attn_raw)   // (B, HEADS, 24, 24)
{
    int bh   = blockIdx.x;
    int wave = threadIdx.x >> 5;
    int l    = threadIdx.x & 31;
    int c0   = (wave >> 1) * 16;
    int d0   = (wave & 1) * 16;
    int m    = l & 15, g = l >> 4;

    const bf16_t* q1r = q1n + ((size_t)bh * CHP + c0 + m) * NPIX;
    const bf16_t* q2r = q2n + ((size_t)bh * CHP + c0 + m) * NPIX;
    const bf16_t* kyr = kyn + ((size_t)bh * CHP + d0 + m) * NPIX;
    const bf16_t* kxr = kxn + ((size_t)bh * CHP + d0 + m) * NPIX;

    v8f acc = {};
#pragma unroll 2
    for (int n0 = 0; n0 < NPIX; n0 += 32) {
        // prefetch ~8KB (4096 bf16) ahead (wraps within the padded buffers; always valid)
        int pf = (n0 + 4096) & (NPIX - 1);
        __builtin_prefetch(q1r + pf, 0, 0);
        __builtin_prefetch(q2r + pf, 0, 0);
        __builtin_prefetch(kyr + pf, 0, 0);
        __builtin_prefetch(kxr + pf, 0, 0);

        const v8bf* a1p = (const v8bf*)(q1r + n0);
        const v8bf* a2p = (const v8bf*)(q2r + n0);
        v16bf a1 = CAT16(a1p[2 * g], a1p[2 * g + 1]);
        v16bf a2 = CAT16(a2p[2 * g], a2p[2 * g + 1]);
        v16bf bk = *(const v16bf*)(kyr + n0 + 16 * g);
        v16bf bx = *(const v16bf*)(kxr + n0 + 16 * g);
        acc = __builtin_amdgcn_wmma_f32_16x16x32_bf16(
            false, a1, false, bk, (short)0, acc, false, false);
        acc = __builtin_amdgcn_wmma_f32_16x16x32_bf16(
            false, a2, false, bx, (short)0, acc, false, false);
    }
#pragma unroll
    for (int r = 0; r < 8; ++r) {
        int c = c0 + r + 8 * g;
        int d = d0 + m;
        if (c < CH && d < CH)
            attn_raw[((size_t)bh * CH + c) * CH + d] = acc[r];
    }
}

// ---------------------------------------------------------------------------
// 9) temperature * scores -> softmax over d (24) -> padded 32x32 bf16
// ---------------------------------------------------------------------------
__global__ void softmax_attn(const float* __restrict__ attn_raw,
                             const float* __restrict__ temperature,
                             bf16_t* __restrict__ attn_bf)  // (B, HEADS, 32, 32)
{
    int rid = blockIdx.x;
    int r = rid % CHP;
    int h = (rid / CHP) % HEADS;
    int b = rid / (CHP * HEADS);
    int l = threadIdx.x;                   // 32 lanes, one wave
    bf16_t* dst = attn_bf + ((size_t)(b * HEADS + h) * CHP + r) * CHP;
    if (r >= CH) { dst[l] = (bf16_t)0.f; return; }
    float t = temperature[h];
    float v = (l < CH)
        ? attn_raw[((size_t)(b * HEADS + h) * CH + r) * CH + l] * t
        : -3.0e38f;
    float mx = v;
    for (int o = 16; o > 0; o >>= 1) mx = fmaxf(mx, __shfl_xor(mx, o, 32));
    float e = (l < CH) ? __expf(v - mx) : 0.f;
    float s = e;
    for (int o = 16; o > 0; o >>= 1) s += __shfl_xor(s, o, 32);
    dst[l] = (bf16_t)(e / s);
}

// ---------------------------------------------------------------------------
// 10) out = gamma1*(attn@vx) + gamma2*(attn@vy); one 16x16 tile per wave,
//     single K=32 step; all operand loads are contiguous 16B/32B.
// ---------------------------------------------------------------------------
__global__ void out_gemm(const bf16_t* __restrict__ attn_bf,
                         const bf16_t* __restrict__ vxT,  // (B*H, NPIX, 32)
                         const bf16_t* __restrict__ vyT,
                         const float* __restrict__ gamma1,
                         const float* __restrict__ gamma2,
                         float* __restrict__ out)         // (B, 192, NPIX)
{
    int wg = blockIdx.x * 4 + (threadIdx.x >> 5);
    int l  = threadIdx.x & 31;
    int nt = wg & 1023;
    int ct = (wg >> 10) & 1;
    int bh = wg >> 11;
    int h = bh & 7, b = bh >> 3;
    int c0 = ct * 16, n0 = nt * 16;
    int m = l & 15, g = l >> 4;

    const v8bf* ap = (const v8bf*)(attn_bf + ((size_t)bh * CHP + c0 + m) * CHP);
    v16bf a = CAT16(ap[2 * g], ap[2 * g + 1]);

    const bf16_t* vxc = vxT + ((size_t)bh * NPIX + n0 + m) * CHP;
    const bf16_t* vyc = vyT + ((size_t)bh * NPIX + n0 + m) * CHP;
    v16bf bx = *(const v16bf*)(vxc + 16 * g);
    v16bf by = *(const v16bf*)(vyc + 16 * g);

    v8f accx = {}, accy = {};
    accx = __builtin_amdgcn_wmma_f32_16x16x32_bf16(
        false, a, false, bx, (short)0, accx, false, false);
    accy = __builtin_amdgcn_wmma_f32_16x16x32_bf16(
        false, a, false, by, (short)0, accy, false, false);

    float g1 = gamma1[0], g2 = gamma2[0];
#pragma unroll
    for (int r = 0; r < 8; ++r) {
        int c = c0 + r + 8 * g;
        if (c < CH)
            out[((size_t)b * CIN + h * CH + c) * NPIX + n0 + m]
                = g1 * accx[r] + g2 * accy[r];
    }
}

// ---------------------------------------------------------------------------
// Launch
// ---------------------------------------------------------------------------
extern "C" void kernel_launch(void* const* d_in, const int* in_sizes, int n_in,
                              void* d_out, int out_size, void* d_ws, size_t ws_size,
                              hipStream_t stream)
{
    (void)in_sizes; (void)n_in; (void)out_size; (void)ws_size;
    const float* x       = (const float*)d_in[0];
    const float* y       = (const float*)d_in[1];
    const float* kv_w    = (const float*)d_in[2];
    const float* kv_b    = (const float*)d_in[3];
    const float* kv_dw_w = (const float*)d_in[4];
    const float* kv_dw_b = (const float*)d_in[5];
    const float* q_w     = (const float*)d_in[6];
    const float* q_b     = (const float*)d_in[7];
    const float* q_dw_w  = (const float*)d_in[8];
    const float* q_dw_b  = (const float*)d_in[9];
    const float* ky_w    = (const float*)d_in[10];
    const float* ky_b    = (const float*)d_in[11];
    const float* ky_dw_w = (const float*)d_in[12];
    const float* ky_dw_b = (const float*)d_in[13];
    const float* temperature = (const float*)d_in[14];
    const float* gamma1  = (const float*)d_in[15];
    const float* gamma2  = (const float*)d_in[16];
    float* out = (float*)d_out;

    char* ws = (char*)d_ws;
    size_t off = 0;
    auto alloc = [&](size_t bytes) {
        size_t o = off;
        off += (bytes + 255) & ~(size_t)255;
        return o;
    };

    const size_t in_bf   = (size_t)BATCH * CIN * NPIX * sizeof(bf16_t);
    const size_t conv_f  = (size_t)BATCH * CC2 * NPIX * sizeof(float);
    const size_t pack_bf = (size_t)BATCH * HEADS * CHP * NPIX * sizeof(bf16_t);

    bf16_t* wbf  = (bf16_t*)(ws + alloc(3ull * CC2 * CIN * sizeof(bf16_t)));
    bf16_t* xT   = (bf16_t*)(ws + alloc(in_bf));
    bf16_t* yT   = (bf16_t*)(ws + alloc(in_bf));
    bf16_t* xmT  = (bf16_t*)(ws + alloc(in_bf));
    float* conv_tmp = (float*)(ws + alloc(conv_f));
    float* gq   = (float*)(ws + alloc(conv_f));
    float* gkvx = (float*)(ws + alloc(conv_f));
    float* gkvy = (float*)(ws + alloc(conv_f));
    float* invn = (float*)(ws + alloc(3ull * BATCH * CC2 * sizeof(float)));
    bf16_t* q1n = (bf16_t*)(ws + alloc(pack_bf));
    bf16_t* q2n = (bf16_t*)(ws + alloc(pack_bf));
    bf16_t* kxn = (bf16_t*)(ws + alloc(pack_bf));
    bf16_t* kyn = (bf16_t*)(ws + alloc(pack_bf));
    bf16_t* vxT = (bf16_t*)(ws + alloc(pack_bf));
    bf16_t* vyT = (bf16_t*)(ws + alloc(pack_bf));
    float*  attn_raw = (float*)(ws + alloc((size_t)BATCH * HEADS * CH * CH * sizeof(float)));
    bf16_t* attn_bf  = (bf16_t*)(ws + alloc((size_t)BATCH * HEADS * CHP * CHP * sizeof(bf16_t)));

    // 1) downcasts (inputs transposed for contiguous WMMA-B loads)
    cvt_weights<<<(3 * CC2 * CIN + 255) / 256, 256, 0, stream>>>(kv_w, q_w, ky_w, wbf);
    {
        dim3 tgrid(NPIX / 32, CIN / 32, BATCH);
        cvt_inputs_t<<<tgrid, 256, 0, stream>>>(x, y, xT, yT, xmT);
    }

    // 2+3) three branches: conv1x1 (WMMA) -> grouped 3x3 conv (f32)
    dim3 cgrid(NPIX / 128, CC2 / 16, BATCH);
    int ggrid = (int)(((size_t)BATCH * CC2 * NPIX) / 256);

    conv1x1_wmma<<<cgrid, 256, 0, stream>>>(xmT, wbf + 0 * CC2 * CIN, kv_b, conv_tmp);
    gconv3<<<ggrid, 256, 0, stream>>>(conv_tmp, kv_dw_w, kv_dw_b, gq);

    conv1x1_wmma<<<cgrid, 256, 0, stream>>>(xT, wbf + 1 * CC2 * CIN, q_b, conv_tmp);
    gconv3<<<ggrid, 256, 0, stream>>>(conv_tmp, q_dw_w, q_dw_b, gkvx);

    conv1x1_wmma<<<cgrid, 256, 0, stream>>>(yT, wbf + 2 * CC2 * CIN, ky_b, conv_tmp);
    gconv3<<<ggrid, 256, 0, stream>>>(conv_tmp, ky_dw_w, ky_dw_b, gkvy);

    // 4) norms, 5) packing (normalized q/k row-major; v transposed)
    rownorm<<<3 * BATCH * CC2, 256, 0, stream>>>(gq, gkvx, gkvy, invn);
    pack_qk<<<(int)((4ull * BATCH * HEADS * CHP * NPIX) / 256), 256, 0, stream>>>(
        gq, gkvx, gkvy, invn, q1n, q2n, kxn, kyn);
    {
        dim3 vgrid(NPIX / 32, BATCH * HEADS);
        pack_v_t<<<vgrid, 256, 0, stream>>>(gkvx, gkvy, vxT, vyT);
    }

    // 6) attention scores, 7) softmax, 8) output GEMM
    attn_gemm<<<BATCH * HEADS, 128, 0, stream>>>(q1n, q2n, kxn, kyn, attn_raw);
    softmax_attn<<<BATCH * HEADS * CHP, 32, 0, stream>>>(attn_raw, temperature, attn_bf);
    out_gemm<<<(BATCH * HEADS * 2 * (NPIX / 16)) / 4, 128, 0, stream>>>(
        attn_bf, vxT, vyT, gamma1, gamma2, out);
}
